// SelfAttention_10496900071590
// MI455X (gfx1250) — compile-verified, matrix operand order checked by reading, not run on hardware
//
#include <hip/hip_runtime.h>
#include <hip/hip_bf16.h>

#define CH   512
#define NTOK 2304
#define NB   16
#define WSZ  (CH * CH)          // one weight matrix, elements

typedef __attribute__((ext_vector_type(16))) _Float16 v16h;
typedef __attribute__((ext_vector_type(8)))  _Float16 v8h;
typedef __attribute__((ext_vector_type(8)))  float    v8f;

static __device__ inline v16h cat16(v8h lo, v8h hi) {
  return __builtin_shufflevector(lo, hi, 0,1,2,3,4,5,6,7,8,9,10,11,12,13,14,15);
}

static __device__ inline v8f wmma_f16(v16h a, v16h b, v8f c) {
  // D = A(16x32 f16) x B(32x16 f16) + C(16x16 f32)
  return __builtin_amdgcn_wmma_f32_16x16x32_f16(false, a, false, b, (short)0, c,
                                                false, false);
}

// ---------------------------------------------------------------------------
// Kernel 0a: f32 -> f16 convert (weights; one-time)
// ---------------------------------------------------------------------------
__global__ __launch_bounds__(256) void cvt_kernel(const float* __restrict__ s,
                                                  _Float16* __restrict__ d, int n) {
  const int i = blockIdx.x * 256 + threadIdx.x;
  if (i < n) d[i] = (_Float16)s[i];
}

// ---------------------------------------------------------------------------
// Kernel 0b: x [B][CH][NTOK] f32 -> xt [B][NTOK][CH] f16 (LDS-tiled transpose)
// ---------------------------------------------------------------------------
__global__ __launch_bounds__(256) void xpose_kernel(const float* __restrict__ x,
                                                    _Float16* __restrict__ xt) {
  __shared__ float tile[32][33];
  const int b  = blockIdx.z;
  const int c0 = blockIdx.y * 32;
  const int n0 = blockIdx.x * 32;
  const int tx = threadIdx.x & 31;
  const int ty = threadIdx.x >> 5;                  // 8 rows per pass

  const float* xb = x + ((size_t)b * CH + c0) * NTOK + n0;
#pragma unroll
  for (int i = ty; i < 32; i += 8)
    tile[i][tx] = xb[(size_t)i * NTOK + tx];        // coalesced along tokens
  __syncthreads();
  _Float16* xtb = xt + ((size_t)b * NTOK + n0) * CH + c0;
#pragma unroll
  for (int i = ty; i < 32; i += 8)
    xtb[(size_t)i * CH + tx] = (_Float16)tile[tx][i];  // coalesced along channels
}

// ---------------------------------------------------------------------------
// Kernel 1: q/k/v = xt @ W^T + b   (per wave: 16 tokens x all 512 channels)
//   xt : [B][NTOK][CH] f16  -> A frags are contiguous b128 loads
//   wf : [3][CH][CH]   f16  (q,k,v; [N][K] row-major = B-operand lane layout)
//   qf : [B][NTOK][CH] f16, pre-scaled by 1/sqrt(CH)
//   kf : [B][NTOK][CH] f16
//   vt : [B][CH][NTOK] f16  (transposed; packed v8h stores)
// ---------------------------------------------------------------------------
__global__ __launch_bounds__(128) void qkv_kernel(
    const _Float16* __restrict__ xt, const _Float16* __restrict__ wf,
    const float* __restrict__ bq, const float* __restrict__ bk,
    const float* __restrict__ bv,
    _Float16* __restrict__ qf, _Float16* __restrict__ kf,
    _Float16* __restrict__ vt)
{
  const int lane  = threadIdx.x & 31;
  const int wave  = threadIdx.x >> 5;
  const int gwave = blockIdx.x * 4 + wave;
  const int b     = gwave / (NTOK / 16);
  const int n0    = (gwave % (NTOK / 16)) * 16;
  const int half  = lane >> 4;     // K-half select per CDNA5 16-bit A/B layout
  const int lr    = lane & 15;

  const _Float16* arow = xt + ((size_t)b * NTOK + n0 + lr) * CH;
  v16h afrag[16];
#pragma unroll
  for (int kk = 0; kk < 16; ++kk) {
    const int kb = kk * 32 + half * 8;
    afrag[kk] = cat16(*(const v8h*)(arow + kb), *(const v8h*)(arow + kb + 16));
  }

  const float qscale = 0.044194173824159216f;  // 1/sqrt(512)

#pragma unroll 1
  for (int which = 0; which < 3; ++which) {
    const _Float16* W = wf + (size_t)which * WSZ;
    const float* bias = (which == 0) ? bq : (which == 1) ? bk : bv;
#pragma unroll 1
    for (int ct = 0; ct < CH / 16; ++ct) {
      const int col = ct * 16 + lr;            // B-operand: lane holds column N
      const _Float16* wrow = W + (size_t)col * CH;
      v8f acc = {};
#pragma unroll
      for (int kk = 0; kk < 16; ++kk) {
        const int kb = kk * 32 + half * 8;
        v16h bf = cat16(*(const v8h*)(wrow + kb), *(const v8h*)(wrow + kb + 16));
        acc = wmma_f16(afrag[kk], bf, acc);
      }
      const float bias_c = bias[col];
      if (which == 2) {
        // pack 8 consecutive token rows -> one 16B store in vt
        v8h pk;
#pragma unroll
        for (int v = 0; v < 8; ++v) pk[v] = (_Float16)(acc[v] + bias_c);
        *(v8h*)(vt + ((size_t)b * CH + col) * NTOK + n0 + half * 8) = pk;
      } else {
#pragma unroll
        for (int v = 0; v < 8; ++v) {
          const int r = n0 + v + half * 8;
          const float val = acc[v] + bias_c;
          if (which == 0)
            qf[((size_t)b * NTOK + r) * CH + col] = (_Float16)(val * qscale);
          else
            kf[((size_t)b * NTOK + r) * CH + col] = (_Float16)val;
        }
      }
    }
  }
}

// ---------------------------------------------------------------------------
// Kernel 2: flash attention. WG = 16 query rows; wave w owns key-slice (S)
// and channel-slice [128w,128w+128). Softmax parallel: 8 lanes per row.
// ---------------------------------------------------------------------------
__global__ __launch_bounds__(128) void attn_kernel(
    const _Float16* __restrict__ qf, const _Float16* __restrict__ kf,
    const _Float16* __restrict__ vt, _Float16* __restrict__ attnout)
{
  __shared__ _Float16 q_lds[16][CH];        // 16 KB: staged query tile
  __shared__ float    s_lds[16][132];       // padded: break bank conflicts
  __shared__ _Float16 p_lds[16][136];       // padded; 272B rows keep 16B align
  __shared__ float    alpha_lds[16];
  __shared__ float    l_lds[16];

  const int lane = threadIdx.x & 31;
  const int wave = threadIdx.x >> 5;
  const int b    = blockIdx.x / (NTOK / 16);
  const int n0   = (blockIdx.x % (NTOK / 16)) * 16;
  const int half = lane >> 4;
  const int lr   = lane & 15;

  // Stage q tile with 16B vector copies
  {
    const uint4* qb = (const uint4*)(qf + ((size_t)b * NTOK + n0) * CH);
    uint4* ql = (uint4*)&q_lds[0][0];
#pragma unroll
    for (int i = threadIdx.x; i < 16 * CH / 8; i += 128) ql[i] = qb[i];
  }
  __syncthreads();

  v8f acc[8];
  {
    v8f z = {};
#pragma unroll
    for (int t = 0; t < 8; ++t) acc[t] = z;
  }

  // softmax state: row r = wave*4 + (lane>>3), replicated across its 8 lanes
  float m_run = -1e30f, l_run = 0.0f;
  const int g   = lane >> 3;
  const int sub = lane & 7;
  const int srow = wave * 4 + g;

  for (int j0 = 0; j0 < NTOK; j0 += 128) {
    // hint-prefetch next key/value block (global_prefetch_b8)
    if (j0 + 128 < NTOK) {
      __builtin_prefetch(kf + ((size_t)b * NTOK + j0 + 128 + threadIdx.x) * CH, 0, 1);
      __builtin_prefetch(vt + ((size_t)b * CH + wave * 128 + lane) * NTOK + j0 + 128, 0, 1);
    }

    // ---- S phase: wave w computes keys [j0+32w, j0+32w+32)
#pragma unroll
    for (int st = 0; st < 2; ++st) {
      const int key = j0 + wave * 32 + st * 16 + lr;
      const _Float16* krow = kf + ((size_t)b * NTOK + key) * CH;
      v8f sc = {};
#pragma unroll
      for (int kk = 0; kk < 16; ++kk) {
        const int kb = kk * 32 + half * 8;
        v16h a = cat16(*(const v8h*)&q_lds[lr][kb],
                       *(const v8h*)&q_lds[lr][kb + 16]);
        v16h bb = cat16(*(const v8h*)(krow + kb),
                        *(const v8h*)(krow + kb + 16));
        sc = wmma_f16(a, bb, sc);
      }
#pragma unroll
      for (int v = 0; v < 8; ++v)
        s_lds[v + 8 * half][wave * 32 + st * 16 + lr] = sc[v];
    }
    __syncthreads();

    // ---- online softmax: every row handled by an 8-lane group
    {
      float mb = -1e30f;
#pragma unroll
      for (int jj = 0; jj < 16; ++jj)
        mb = fmaxf(mb, s_lds[srow][sub + jj * 8]);
#pragma unroll
      for (int off = 1; off < 8; off <<= 1)
        mb = fmaxf(mb, __shfl_xor(mb, off, 8));
      const float m_new = fmaxf(m_run, mb);
      const float alpha = __expf(m_run - m_new);
      float ls = 0.0f;
#pragma unroll
      for (int jj = 0; jj < 16; ++jj) {
        const int j = sub + jj * 8;
        const float p = __expf(s_lds[srow][j] - m_new);
        p_lds[srow][j] = (_Float16)p;
        ls += p;
      }
#pragma unroll
      for (int off = 1; off < 8; off <<= 1)
        ls += __shfl_xor(ls, off, 8);
      l_run = l_run * alpha + ls;
      m_run = m_new;
      if (sub == 0) alpha_lds[srow] = alpha;
    }
    __syncthreads();

    // ---- rescale accumulators by alpha(row)
#pragma unroll
    for (int v = 0; v < 8; ++v) {
      const float al = alpha_lds[v + 8 * half];
#pragma unroll
      for (int t = 0; t < 8; ++t) acc[t][v] *= al;
    }

    // ---- PV: A = P (16x128 keys), B = vt channel slice of this wave
    v16h pf[4];
#pragma unroll
    for (int kk = 0; kk < 4; ++kk) {
      const int kb = kk * 32 + half * 8;
      pf[kk] = cat16(*(const v8h*)&p_lds[lr][kb],
                     *(const v8h*)&p_lds[lr][kb + 16]);
    }
#pragma unroll
    for (int ct = 0; ct < 8; ++ct) {
      const int c = wave * 128 + ct * 16 + lr;
      const _Float16* vrow = vt + ((size_t)b * CH + c) * NTOK + j0;
#pragma unroll
      for (int kk = 0; kk < 4; ++kk) {
        const int kb = kk * 32 + half * 8;
        v16h bb = cat16(*(const v8h*)(vrow + kb),
                        *(const v8h*)(vrow + kb + 16));
        acc[ct] = wmma_f16(pf[kk], bb, acc[ct]);
      }
    }
    __syncthreads();   // protect s_lds/p_lds before next block
  }

  if (sub == 0) l_lds[srow] = l_run;
  __syncthreads();

  // normalize and store attnout [B][NTOK][CH] f16
#pragma unroll
  for (int ct = 0; ct < 8; ++ct) {
    const int c = wave * 128 + ct * 16 + lr;
#pragma unroll
    for (int v = 0; v < 8; ++v) {
      const int r = n0 + v + 8 * half;
      const float inv = 1.0f / l_lds[v + 8 * half];
      attnout[((size_t)b * NTOK + r) * CH + c] = (_Float16)(acc[ct][v] * inv);
    }
  }
}

// ---------------------------------------------------------------------------
// Kernel 3: out = attnout @ Wp^T + bp + x, stored fp32 in (b, c, h*w) layout
// ---------------------------------------------------------------------------
__global__ __launch_bounds__(128) void proj_kernel(
    const _Float16* __restrict__ ao, const _Float16* __restrict__ wp,
    const float* __restrict__ bp, const float* __restrict__ x,
    float* __restrict__ out)
{
  const int lane  = threadIdx.x & 31;
  const int wave  = threadIdx.x >> 5;
  const int gwave = blockIdx.x * 4 + wave;
  const int b     = gwave / (NTOK / 16);
  const int n0    = (gwave % (NTOK / 16)) * 16;
  const int half  = lane >> 4;
  const int lr    = lane & 15;

  const _Float16* arow = ao + ((size_t)b * NTOK + n0 + lr) * CH;
  v16h afrag[16];
#pragma unroll
  for (int kk = 0; kk < 16; ++kk) {
    const int kb = kk * 32 + half * 8;
    afrag[kk] = cat16(*(const v8h*)(arow + kb), *(const v8h*)(arow + kb + 16));
  }

#pragma unroll 1
  for (int ct = 0; ct < CH / 16; ++ct) {
    const int col = ct * 16 + lr;
    const _Float16* wrow = wp + (size_t)col * CH;
    v8f acc = {};
#pragma unroll
    for (int kk = 0; kk < 16; ++kk) {
      const int kb = kk * 32 + half * 8;
      v16h bf = cat16(*(const v8h*)(wrow + kb), *(const v8h*)(wrow + kb + 16));
      acc = wmma_f16(afrag[kk], bf, acc);
    }
    const float bias_c = bp[col];
    // residual + store: contiguous along tokens within this lane -> float4 x2
    const size_t base = ((size_t)b * CH + col) * NTOK + n0 + half * 8;
    const float4 x0 = *(const float4*)(x + base);
    const float4 x1 = *(const float4*)(x + base + 4);
    float4 o0, o1;
    o0.x = x0.x + acc[0] + bias_c;  o0.y = x0.y + acc[1] + bias_c;
    o0.z = x0.z + acc[2] + bias_c;  o0.w = x0.w + acc[3] + bias_c;
    o1.x = x1.x + acc[4] + bias_c;  o1.y = x1.y + acc[5] + bias_c;
    o1.z = x1.z + acc[6] + bias_c;  o1.w = x1.w + acc[7] + bias_c;
    *(float4*)(out + base)     = o0;
    *(float4*)(out + base + 4) = o1;
  }
}

// ---------------------------------------------------------------------------
extern "C" void kernel_launch(void* const* d_in, const int* in_sizes, int n_in,
                              void* d_out, int out_size, void* d_ws, size_t ws_size,
                              hipStream_t stream) {
  const float* x  = (const float*)d_in[0];
  const float* Wq = (const float*)d_in[1];
  const float* bq = (const float*)d_in[2];
  const float* Wk = (const float*)d_in[3];
  const float* bk = (const float*)d_in[4];
  const float* Wv = (const float*)d_in[5];
  const float* bv = (const float*)d_in[6];
  const float* Wp = (const float*)d_in[7];
  const float* bp = (const float*)d_in[8];
  float* out = (float*)d_out;

  const size_t per = (size_t)NB * NTOK * CH;     // elements per f16 tensor
  _Float16* qf   = (_Float16*)d_ws;
  _Float16* kf   = qf + per;
  _Float16* vt   = kf + per;
  _Float16* xtao = vt + per;   // xt for kernel 1, reused as attn-out after
  _Float16* wf   = xtao + per; // [4][CH][CH] f16: Wq, Wk, Wv, Wp

  // one-time conversions
  cvt_kernel<<<WSZ / 256, 256, 0, stream>>>(Wq, wf + 0 * (size_t)WSZ, WSZ);
  cvt_kernel<<<WSZ / 256, 256, 0, stream>>>(Wk, wf + 1 * (size_t)WSZ, WSZ);
  cvt_kernel<<<WSZ / 256, 256, 0, stream>>>(Wv, wf + 2 * (size_t)WSZ, WSZ);
  cvt_kernel<<<WSZ / 256, 256, 0, stream>>>(Wp, wf + 3 * (size_t)WSZ, WSZ);
  xpose_kernel<<<dim3(NTOK / 32, CH / 32, NB), 256, 0, stream>>>(x, xtao);

  // 16 batches * 144 row-tiles = 2304 waves; 4 waves per 128-thread block
  qkv_kernel<<<NB * (NTOK / 16) / 4, 128, 0, stream>>>(xtao, wf, bq, bk, bv,
                                                       qf, kf, vt);
  attn_kernel<<<NB * (NTOK / 16), 128, 0, stream>>>(qf, kf, vt, xtao);
  proj_kernel<<<NB * (NTOK / 16) / 4, 128, 0, stream>>>(xtao, wf + 3 * (size_t)WSZ,
                                                        bp, x, out);
}
